// NonCausalSelfAttention_22376779612373
// MI455X (gfx1250) — matrix operand order repaired and checked by reading, not compiled, not run on hardware
//
#include <hip/hip_runtime.h>

typedef __attribute__((ext_vector_type(16))) __bf16 v16bf;
typedef __attribute__((ext_vector_type(8)))  float  v8f;
typedef int v4i_g __attribute__((vector_size(16)));   // int4 for async-LDS builtin

union FragB { unsigned int u[8]; v16bf v; };

__device__ __forceinline__ unsigned short f2bf(float f) {
    unsigned int u = __builtin_bit_cast(unsigned int, f);
    unsigned int r = u + 0x7FFFu + ((u >> 16) & 1u);   // RNE
    return (unsigned short)(r >> 16);
}
__device__ __forceinline__ unsigned int f2bf_pack(float lo, float hi) {
    return (unsigned int)f2bf(lo) | ((unsigned int)f2bf(hi) << 16);
}
__device__ __forceinline__ float bf2f(unsigned short h) {
    unsigned int u = ((unsigned int)h) << 16;
    return __builtin_bit_cast(float, u);
}

// CDNA5 async global->LDS copy (ASYNCcnt path), with compile-safe fallback.
#if defined(__gfx1250__) && __has_builtin(__builtin_amdgcn_global_load_async_to_lds_b128)
#define USE_ASYNC_LDS 1
#else
#define USE_ASYNC_LDS 0
#endif

#if USE_ASYNC_LDS
__device__ __forceinline__ void async_copy_b128(const void* gsrc, void* ldst) {
    __builtin_amdgcn_global_load_async_to_lds_b128(
        (__attribute__((address_space(1))) v4i_g*)(uintptr_t)gsrc,
        (__attribute__((address_space(3))) v4i_g*)(unsigned)(uintptr_t)ldst,
        0, 0);
}
__device__ __forceinline__ void wait_async0() {
#if __has_builtin(__builtin_amdgcn_s_wait_asynccnt)
    __builtin_amdgcn_s_wait_asynccnt(0);
#else
    asm volatile("s_wait_asynccnt 0" ::: "memory");
#endif
}
#endif

// ---------------------------------------------------------------------------
// Generic tiled GEMM: C[M,N] = A[M,K] * B[K,N]
// A: fp32 or bf16 (template), B: fp32 (converted to bf16 in staging),
// C: fp32 or bf16 (template). 128x128 tile / 256 threads (8 waves).
// Wave w computes rows [w*16, w*16+16) x all 128 cols (8 WMMA subtiles).
// ---------------------------------------------------------------------------
template <bool A_BF16, bool OUT_BF16>
__global__ __launch_bounds__(256) void gemm_bf16_wmma(
    const void* __restrict__ Av, const float* __restrict__ B,
    void* __restrict__ Cv, int M, int N, int K)
{
    __shared__ __attribute__((aligned(16))) unsigned short lds_a[128 * 32]; // [m][k]
    __shared__ __attribute__((aligned(16))) unsigned short lds_b[128 * 32]; // [n][k]

    const int tid  = threadIdx.x;
    const int lane = tid & 31;
    const int wave = tid >> 5;
    const int half = lane >> 4;
    const int ln   = lane & 15;
    const int m0   = blockIdx.y * 128;
    const int n0   = blockIdx.x * 128;

    v8f acc[8] = {};

    for (int k0 = 0; k0 < K; k0 += 32) {
        __syncthreads();
        // stage A tile (128x32) as bf16, packed dword stores (pairs along K)
        #pragma unroll
        for (int it = 0; it < 8; ++it) {
            int idx = tid + it * 256;            // pair index: 2048 pairs
            int r = idx >> 4, c2 = idx & 15;     // k = 2*c2
            unsigned int pack;
            if (A_BF16) {
                pack = *(const unsigned int*)((const unsigned short*)Av +
                                              (size_t)(m0 + r) * K + k0 + 2 * c2);
            } else {
                const float* ap = (const float*)Av + (size_t)(m0 + r) * K + k0 + 2 * c2;
                pack = f2bf_pack(ap[0], ap[1]);
            }
            ((unsigned int*)lds_a)[idx] = pack;  // lds_a[r*32 + 2*c2]
        }
        // stage B tile (32x128) transposed -> [n][k], packed pairs along K
        #pragma unroll
        for (int it = 0; it < 8; ++it) {
            int idx = tid + it * 256;            // 2048 pairs
            int n = idx & 127, k2 = idx >> 7;    // kk = 2*k2
            const float* bp = B + (size_t)(k0 + 2 * k2) * N + n0 + n;
            ((unsigned int*)lds_b)[n * 16 + k2] = f2bf_pack(bp[0], bp[N]);
        }
        // prefetch next K tile while this one is consumed
        if (k0 + 32 < K) {
            if (A_BF16)
                __builtin_prefetch((const unsigned short*)Av + (size_t)(m0 + (tid >> 1)) * K + k0 + 32, 0, 0);
            else
                __builtin_prefetch((const float*)Av + (size_t)(m0 + (tid >> 1)) * K + k0 + 32, 0, 0);
            __builtin_prefetch(B + (size_t)(k0 + 32 + (tid >> 7)) * N + n0 + (tid & 127), 0, 0);
        }
        __syncthreads();

        // A fragment (16x32): lane holds row ln, half selects K {+0,+8} per group
        FragB af;
        #pragma unroll
        for (int v = 0; v < 8; ++v) {
            int k = (v < 4) ? (2 * v + 8 * half) : (16 + 2 * (v - 4) + 8 * half);
            af.u[v] = *(const unsigned int*)&lds_a[(wave * 16 + ln) * 32 + k];
        }
        #pragma unroll
        for (int nn = 0; nn < 8; ++nn) {
            // B fragment (32x16): lane holds col ln, half selects K {+0,+8} per group
            FragB bfr;
            #pragma unroll
            for (int v = 0; v < 8; ++v)
                bfr.u[v] = *(const unsigned int*)&lds_b[(nn * 16 + ln) * 32 + ((v < 4) ? (2 * v + 8 * half) : (16 + 2 * (v - 4) + 8 * half))];
            acc[nn] = __builtin_amdgcn_wmma_f32_16x16x32_bf16(
                false, af.v, false, bfr.v, (short)0, acc[nn], false, false);
        }
    }

    // epilogue: D layout row = r + 8*half, col = ln
    #pragma unroll
    for (int nn = 0; nn < 8; ++nn) {
        #pragma unroll
        for (int r = 0; r < 8; ++r) {
            size_t row = (size_t)(m0 + wave * 16 + r + 8 * half);
            int    col = n0 + nn * 16 + ln;
            float  val = acc[nn][r];
            if (OUT_BF16) ((unsigned short*)Cv)[row * N + col] = f2bf(val);
            else          ((float*)Cv)[row * N + col] = val;
        }
    }
}

// ---------------------------------------------------------------------------
// RoPE + head split: qkv bf16 [B,S,3D] -> Q (scaled, roped), K (roped), V
// each as bf16 [B,H,S,hd]. One thread per (b,s,h,d2), d2 in [0,64).
// ---------------------------------------------------------------------------
__global__ __launch_bounds__(256) void rope_split_kernel(
    const unsigned short* __restrict__ qkv,
    unsigned short* __restrict__ Q,
    unsigned short* __restrict__ Kd,
    unsigned short* __restrict__ Vd)
{
    const int D = 2048, hd = 128, S = 2048, H = 16;
    long long idx = (long long)blockIdx.x * blockDim.x + threadIdx.x;
    int d2 = (int)(idx & 63); long long t = idx >> 6;
    int h  = (int)(t & 15);  t >>= 4;
    int s  = (int)(t & 2047);
    int b  = (int)(t >> 11);

    const float scale = 0.08838834764831845f; // hd^-0.5

    size_t base = ((size_t)(b * S + s)) * 3 * D;
    const unsigned short* qp = qkv + base + h * hd;
    const unsigned short* kp = qkv + base + D + h * hd;
    const unsigned short* vp = qkv + base + 2 * D + h * hd;

    float inv = __powf(10000.0f, -(float)(2 * d2) / 128.0f);
    float ang = (float)s * inv;
    float sn, cs;
    __sincosf(ang, &sn, &cs);

    float q1 = bf2f(qp[d2]), q2 = bf2f(qp[d2 + 64]);
    float k1 = bf2f(kp[d2]), k2 = bf2f(kp[d2 + 64]);
    float qr1 = q1 * cs - q2 * sn, qr2 = q2 * cs + q1 * sn;
    float kr1 = k1 * cs - k2 * sn, kr2 = k2 * cs + k1 * sn;

    size_t ob = (((size_t)(b * H + h)) * S + s) * hd;
    Q[ob + d2]       = f2bf(qr1 * scale);
    Q[ob + d2 + 64]  = f2bf(qr2 * scale);
    Kd[ob + d2]      = f2bf(kr1);
    Kd[ob + d2 + 64] = f2bf(kr2);
    Vd[ob + d2]      = vp[d2];
    Vd[ob + d2 + 64] = vp[d2 + 64];
}

// ---------------------------------------------------------------------------
// Flash attention (non-causal). One WG per (b,h,128 query rows); 8 waves,
// wave w owns query rows [qr0, qr0+16). K/V staged in LDS 32 keys at a time.
// K tile is a contiguous 8KB copy -> async global->LDS DMA on CDNA5.
// ---------------------------------------------------------------------------
__global__ __launch_bounds__(256) void flash_attn_kernel(
    const unsigned short* __restrict__ Q,
    const unsigned short* __restrict__ Kd,
    const unsigned short* __restrict__ Vd,
    unsigned short* __restrict__ O)
{
    const int S = 2048, hd = 128, D = 2048;
    __shared__ __attribute__((aligned(16))) unsigned short lds_k[32 * 128];  // [key][d]
    __shared__ __attribute__((aligned(16))) unsigned short lds_v[128 * 32];  // [d][key]
    __shared__ __attribute__((aligned(16))) unsigned short lds_p[8][16 * 32];// per-wave [row][key]

    const int tid  = threadIdx.x;
    const int lane = tid & 31;
    const int wave = tid >> 5;
    const int half = lane >> 4;
    const int ln   = lane & 15;

    const int wg = blockIdx.x;          // 512 = 32 (b,h) * 16 query blocks
    const int qb = wg & 15;
    const int bh = wg >> 4;
    const int b  = bh >> 4, h = bh & 15;
    const size_t kv_base = (size_t)bh * S * hd;
    const int qr0 = qb * 128 + wave * 16;

    // Q fragments (16x128 = 4 chunks of 16x32), loaded straight from global
    FragB qf[4];
    #pragma unroll
    for (int c = 0; c < 4; ++c) {
        #pragma unroll
        for (int v = 0; v < 8; ++v) {
            int k = c * 32 + ((v < 4) ? (2 * v + 8 * half) : (16 + 2 * (v - 4) + 8 * half));
            qf[c].u[v] = *(const unsigned int*)(Q + kv_base + (size_t)(qr0 + ln) * hd + k);
        }
    }

    float mrow[8], lrow[8];
    #pragma unroll
    for (int r = 0; r < 8; ++r) { mrow[r] = -1e30f; lrow[r] = 0.0f; }
    v8f accO[8] = {};

    for (int j0 = 0; j0 < S; j0 += 32) {
        __syncthreads();
        // K tile: contiguous 8KB region [j0*hd .. j0*hd + 4096 ushorts)
        {
            const unsigned short* ksrc = Kd + kv_base + (size_t)j0 * hd;
#if USE_ASYNC_LDS
            #pragma unroll
            for (int it = 0; it < 2; ++it) {
                int e = (tid + it * 256) * 8;    // 16B granules
                async_copy_b128(ksrc + e, &lds_k[e]);
            }
#else
            #pragma unroll
            for (int it = 0; it < 2; ++it) {
                int e = (tid + it * 256) * 8;
                *(uint4*)&lds_k[e] = *(const uint4*)(ksrc + e);
            }
#endif
        }
        // V tile transposed -> [d][key], packed dword stores (pairs along key)
        #pragma unroll
        for (int it = 0; it < 8; ++it) {
            int idx = tid + it * 256;            // 2048 pairs
            int d = idx & 127, k2 = idx >> 7;    // keys 2*k2, 2*k2+1
            const unsigned short* vp = Vd + kv_base + (size_t)(j0 + 2 * k2) * hd + d;
            unsigned int pack = (unsigned int)vp[0] | ((unsigned int)vp[hd] << 16);
            ((unsigned int*)lds_v)[d * 16 + k2] = pack;
        }
#if USE_ASYNC_LDS
        wait_async0();
#endif
        __syncthreads();

        // scores: two 16x16 tiles (keys j0..j0+15, j0+16..j0+31)
        v8f sc[2];
        #pragma unroll
        for (int t = 0; t < 2; ++t) {
            v8f s = {};
            #pragma unroll
            for (int c = 0; c < 4; ++c) {
                FragB kf;
                #pragma unroll
                for (int v = 0; v < 8; ++v)
                    kf.u[v] = *(const unsigned int*)&lds_k[(t * 16 + ln) * 128 + c * 32 + ((v < 4) ? (2 * v + 8 * half) : (16 + 2 * (v - 4) + 8 * half))];
                s = __builtin_amdgcn_wmma_f32_16x16x32_bf16(
                    false, qf[c].v, false, kf.v, (short)0, s, false, false);
            }
            sc[t] = s;
        }

        // online softmax: rows r+8*half live across the 16 lanes of each half
        #pragma unroll
        for (int r = 0; r < 8; ++r) {
            float mt = fmaxf(sc[0][r], sc[1][r]);
            #pragma unroll
            for (int off = 1; off < 16; off <<= 1)
                mt = fmaxf(mt, __shfl_xor(mt, off, 32));
            float mnew  = fmaxf(mrow[r], mt);
            float alpha = __expf(mrow[r] - mnew);
            float p0 = __expf(sc[0][r] - mnew);
            float p1 = __expf(sc[1][r] - mnew);
            sc[0][r] = p0; sc[1][r] = p1;
            float ps = p0 + p1;
            #pragma unroll
            for (int off = 1; off < 16; off <<= 1)
                ps += __shfl_xor(ps, off, 32);
            lrow[r] = lrow[r] * alpha + ps;
            mrow[r] = mnew;
            #pragma unroll
            for (int nn = 0; nn < 8; ++nn) accO[nn][r] = accO[nn][r] * alpha;
        }

        // transpose P (D-layout -> A-layout) through wave-private LDS
        #pragma unroll
        for (int r = 0; r < 8; ++r) {
            lds_p[wave][(r + 8 * half) * 32 + ln]      = f2bf(sc[0][r]);
            lds_p[wave][(r + 8 * half) * 32 + 16 + ln] = f2bf(sc[1][r]);
        }
        asm volatile("s_wait_dscnt 0" ::: "memory"); // CDNA5 split DS counter wait

        FragB pf;
        #pragma unroll
        for (int v = 0; v < 8; ++v) {
            int k = (v < 4) ? (2 * v + 8 * half) : (16 + 2 * (v - 4) + 8 * half);
            pf.u[v] = *(const unsigned int*)&lds_p[wave][ln * 32 + k];
        }
        // O += P(16x32) * V(32x128)
        #pragma unroll
        for (int nn = 0; nn < 8; ++nn) {
            FragB vf;
            #pragma unroll
            for (int v = 0; v < 8; ++v)
                vf.u[v] = *(const unsigned int*)&lds_v[(nn * 16 + ln) * 32 + ((v < 4) ? (2 * v + 8 * half) : (16 + 2 * (v - 4) + 8 * half))];
            accO[nn] = __builtin_amdgcn_wmma_f32_16x16x32_bf16(
                false, pf.v, false, vf.v, (short)0, accO[nn], false, false);
        }
    }

    // epilogue: O[b, row, h*hd + col] bf16, normalized by row sum
    #pragma unroll
    for (int nn = 0; nn < 8; ++nn) {
        #pragma unroll
        for (int r = 0; r < 8; ++r) {
            int row = qr0 + r + 8 * half;
            int col = h * hd + nn * 16 + ln;
            O[((size_t)(b * S + row)) * D + col] = f2bf(accO[nn][r] * (1.0f / lrow[r]));
        }
    }
}

// ---------------------------------------------------------------------------
extern "C" void kernel_launch(void* const* d_in, const int* in_sizes, int n_in,
                              void* d_out, int out_size, void* d_ws, size_t ws_size,
                              hipStream_t stream) {
    const float* x     = (const float*)d_in[0];
    const float* w_qkv = (const float*)d_in[1];
    const float* w_out = (const float*)d_in[2];
    float* out = (float*)d_out;

    const int B = 2, S = 2048, D = 2048, H = 16;
    const int M = B * S;

    unsigned short* qkv = (unsigned short*)d_ws;        // [M, 3D] bf16
    unsigned short* Qb  = qkv + (size_t)M * 3 * D;      // [B,H,S,hd] bf16
    unsigned short* Kb  = Qb + (size_t)M * D;
    unsigned short* Vb  = Kb + (size_t)M * D;
    unsigned short* Ob  = Vb + (size_t)M * D;           // [B,S,D] bf16

    dim3 blk(256);

    // 1) qkv = x @ w_qkv  (bf16 out)
    gemm_bf16_wmma<false, true>
        <<<dim3(3 * D / 128, M / 128), blk, 0, stream>>>(x, w_qkv, qkv, M, 3 * D, D);

    // 2) RoPE + split heads (+ fold hd^-0.5 into Q)
    rope_split_kernel<<<dim3((unsigned)((size_t)M * H * 64 / 256)), blk, 0, stream>>>(qkv, Qb, Kb, Vb);

    // 3) fused flash attention
    flash_attn_kernel<<<dim3(B * H * (S / 128)), blk, 0, stream>>>(Qb, Kb, Vb, Ob);

    // 4) out = O @ w_out  (fp32 out)
    gemm_bf16_wmma<true, false>
        <<<dim3(D / 128, M / 128), blk, 0, stream>>>(Ob, w_out, out, M, D, D);
}
